// ZLSTM_3246995276447
// MI455X (gfx1250) — matrix operand order repaired and checked by reading, not compiled, not run on hardware
//
#include <hip/hip_runtime.h>

// Problem sizes (fixed by the reference)
#define HID    1024
#define TT     512
#define BATCH  32
#define FOURH  4096
#define KCAT   2048                 // [x ; h] concatenated K
#define NCHUNK (KCAT / 32)          // 64 K-chunks of 32
#define HSTATE (BATCH * HID)        // 32768 elems, one layer's h
#define PARITY (2 * HSTATE)         // 65536 elems, both layers, one parity

typedef __attribute__((ext_vector_type(16))) _Float16 v16h;
typedef __attribute__((ext_vector_type(8)))  _Float16 v8h;
typedef __attribute__((ext_vector_type(8)))  float    v8f;

__device__ __forceinline__ float sigm(float x) { return 1.0f / (1.0f + __expf(-x)); }

__device__ __forceinline__ v16h cat16(v8h lo, v8h hi) {
  return __builtin_shufflevector(lo, hi, 0,1,2,3,4,5,6,7,8,9,10,11,12,13,14,15);
}

// A fragment: lane covers K = kc+8*lh..+7 and kc+16+8*lh..+7 (pal = row + lh*8)
__device__ __forceinline__ v16h ldA(const _Float16* pal0, const _Float16* pal1, int kc) {
  const _Float16* p = (kc < HID) ? (pal0 + kc) : (pal1 + (kc - HID));
  return cat16(*(const v8h*)p, *(const v8h*)(p + 16));
}
// B fragment: lane covers K = kc+8*lh..+7 and kc+16+8*lh..+7 (pwl = row + lh*8)
__device__ __forceinline__ v16h ldB(const _Float16* pwl, int kc) {
  return cat16(*(const v8h*)(pwl + kc), *(const v8h*)(pwl + kc + 16));
}

#define WMMA(A, Bm, C) \
  __builtin_amdgcn_wmma_f32_16x16x32_f16(false, (A), false, (Bm), (short)0, (C), false, false)

// ---------------------------------------------------------------------------
// Prep kernels
// ---------------------------------------------------------------------------
__global__ void cvt_weights_k(const float* __restrict__ wih,
                              const float* __restrict__ whh,
                              _Float16* __restrict__ w16) {
  size_t i   = (size_t)blockIdx.x * 256 + threadIdx.x;  // [0, 2*4096*2048)
  size_t l   = i >> 23;
  size_t rem = i & ((size_t(1) << 23) - 1);
  size_t n   = rem >> 11;
  size_t k   = rem & 2047;
  size_t src = (l << 22) + (n << 10);
  float v = (k < HID) ? wih[src + k] : whh[src + (k - HID)];
  w16[i] = (_Float16)v;
}

__global__ void cvt_x_k(const float* __restrict__ x, _Float16* __restrict__ x16) {
  size_t i = (size_t)blockIdx.x * 256 + threadIdx.x;     // [0, 32*512*1024)
  x16[i] = (_Float16)x[i];
}

__global__ void prep_state_k(const float* __restrict__ bih,
                             const float* __restrict__ bhh,
                             float* __restrict__ bsum,
                             _Float16* __restrict__ h16,
                             float* __restrict__ h32,
                             float* __restrict__ c32) {
  size_t i = (size_t)blockIdx.x * 256 + threadIdx.x;     // [0, 131072)
  if (i < 2 * FOURH) bsum[i] = bih[i] + bhh[i];
  h16[i] = (_Float16)0.0f;                               // 2 parities * 2 layers * 32k
  if (i < 2 * HSTATE) { h32[i] = 0.0f; c32[i] = 0.0f; }
}

__global__ void tail_k(float* __restrict__ o,
                       const float* __restrict__ h32,
                       const float* __restrict__ c32) {
  size_t i = (size_t)blockIdx.x * 256 + threadIdx.x;     // [0, 131072)
  o[i] = (i < 2 * HSTATE) ? h32[i] : c32[i - 2 * HSTATE];
}

// ---------------------------------------------------------------------------
// One LSTM cell (one layer, one step). Software-pipelined WMMA GEMM (prefetch
// distance = 2 K-chunks, enforced with sched_barrier fences) + fused pointwise.
// Per wave: one 16(batch) x 16(hidden) tile, 4 gate accumulators.
// ---------------------------------------------------------------------------
__device__ __forceinline__ void lstm_cell(
    int tile,                                    // 0..127
    const _Float16* __restrict__ A0, long sA,    // input rows, row stride
    const _Float16* __restrict__ Ap,             // h_prev f16 [32 x 1024]
    const _Float16* __restrict__ W,              // [4096 x 2048] f16
    const float*    __restrict__ bsum,           // [4096]
    float*          __restrict__ c,              // [32 x 1024] f32, in/out
    float*          __restrict__ h32,            // [32 x 1024] f32, out
    _Float16*       __restrict__ h16,            // [32 x 1024] f16, out
    float*          __restrict__ out, long sOut) // optional output slice
{
  const int lane = threadIdx.x & 31;
  const int m0 = (tile & 1) * 16;
  const int j0 = (tile >> 1) * 16;
  const int lm = lane & 15;
  const int lh = lane >> 4;

  // per-lane adjusted pointers
  const _Float16* pal0 = A0 + (size_t)(m0 + lm) * (size_t)sA + lh * 8;
  const _Float16* pal1 = Ap + (size_t)(m0 + lm) * HID + lh * 8;
  const _Float16* pw0 = W + (size_t)(0 * HID + j0 + lm) * KCAT + lh * 8;
  const _Float16* pw1 = W + (size_t)(1 * HID + j0 + lm) * KCAT + lh * 8;
  const _Float16* pw2 = W + (size_t)(2 * HID + j0 + lm) * KCAT + lh * 8;
  const _Float16* pw3 = W + (size_t)(3 * HID + j0 + lm) * KCAT + lh * 8;

  v8f acc0 = {}, acc1 = {}, acc2 = {}, acc3 = {};

  // Double-buffered fragments (constant slot index after unroll -> registers)
  v16h a[2], b0[2], b1[2], b2[2], b3[2];
  a[0]  = ldA(pal0, pal1, 0);
  b0[0] = ldB(pw0, 0); b1[0] = ldB(pw1, 0); b2[0] = ldB(pw2, 0); b3[0] = ldB(pw3, 0);
  a[1]  = ldA(pal0, pal1, 32);
  b0[1] = ldB(pw0, 32); b1[1] = ldB(pw1, 32); b2[1] = ldB(pw2, 32); b3[1] = ldB(pw3, 32);

#pragma unroll
  for (int i = 0; i < NCHUNK; ++i) {
    const int sl = i & 1;
    acc0 = WMMA(a[sl], b0[sl], acc0);
    acc1 = WMMA(a[sl], b1[sl], acc1);
    acc2 = WMMA(a[sl], b2[sl], acc2);
    acc3 = WMMA(a[sl], b3[sl], acc3);
    __builtin_amdgcn_sched_barrier(0);   // keep prefetch below the WMMA group
    if (i < NCHUNK - 2) {
      const int kc = (i + 2) * 32;
      a[sl]  = ldA(pal0, pal1, kc);
      b0[sl] = ldB(pw0, kc);
      b1[sl] = ldB(pw1, kc);
      b2[sl] = ldB(pw2, kc);
      b3[sl] = ldB(pw3, kc);
    }
    __builtin_amdgcn_sched_barrier(0);   // keep next WMMA group below prefetch
  }

  // -------- fused pointwise: lane l, reg r -> (m = m0+8*lh+r, j = j0+lm) ---
  const int j = j0 + lm;
  const float bi = bsum[j];
  const float bf = bsum[HID + j];
  const float bg = bsum[2 * HID + j];
  const float bo = bsum[3 * HID + j];
#pragma unroll
  for (int r = 0; r < 8; ++r) {
    const int m = m0 + lh * 8 + r;
    const size_t idx = (size_t)m * HID + j;
    const float ig = acc0[r] + bi;
    const float fg = acc1[r] + bf;
    const float gg = acc2[r] + bg;
    const float og = acc3[r] + bo;
    const float cv = c[idx];
    const float cn = sigm(fg) * cv + sigm(ig) * tanhf(gg);
    const float hn = sigm(og) * tanhf(cn);
    c[idx]   = cn;
    h32[idx] = hn;
    h16[idx] = (_Float16)hn;
    if (out) out[(size_t)m * (size_t)sOut + j] = hn;
  }
}

// ---------------------------------------------------------------------------
// Diagonal pipelining: one launch computes layer0 @ step t (blocks 0..31)
// and layer1 @ step t-1 (blocks 32..63). Both depend only on the previous
// launch, so no intra-kernel sync is needed. Single cell body; layer selected
// by uniform pointer selects.
// ---------------------------------------------------------------------------
struct LstmArgs {
  const _Float16* x16;   // [B,T,H] f16
  const _Float16* w16;   // [L][4096][2048] f16
  const float*    bsum;  // [L][4096]
  float*          c32;   // [L][32][1024]
  float*          h32;   // [L][32][1024]
  _Float16*       h16;   // [parity][L][32][1024]
  float*          out;   // [B,T,H]
};

__global__ __launch_bounds__(128) void lstm_pair_k(LstmArgs g, int t) {
  const int grp  = blockIdx.x >> 5;                       // 0: layer0, 1: layer1
  const int tile = (blockIdx.x & 31) * 4 + (threadIdx.x >> 5);

  const int step = grp ? (t - 1) : t;
  if (step < 0 || step >= TT) return;

  const size_t pIn  = (size_t)(step & 1) * PARITY;        // own-layer h_prev parity
  const size_t pOut = (size_t)((step + 1) & 1) * PARITY;  // write parity
  const size_t loff = (size_t)grp * HSTATE;

  // layer0 input: x[:, step, :]; layer1 input: h0(step) (previous launch)
  const _Float16* A0 = grp ? (g.h16 + pOut) : (g.x16 + (size_t)step * HID);
  const long      sA = grp ? (long)HID : (long)TT * HID;
  float* outp = grp ? (g.out + (size_t)step * HID) : (float*)nullptr;

  lstm_cell(tile, A0, sA,
            g.h16 + pIn + loff,
            g.w16 + (size_t)grp * FOURH * KCAT,
            g.bsum + (size_t)grp * FOURH,
            g.c32 + loff, g.h32 + loff,
            g.h16 + pOut + loff,
            outp, (long)TT * HID);
}

// ---------------------------------------------------------------------------
extern "C" void kernel_launch(void* const* d_in, const int* in_sizes, int n_in,
                              void* d_out, int out_size, void* d_ws, size_t ws_size,
                              hipStream_t stream) {
  (void)in_sizes; (void)n_in; (void)out_size; (void)ws_size;

  const float* x   = (const float*)d_in[0];
  const float* wih = (const float*)d_in[1];
  const float* whh = (const float*)d_in[2];
  const float* bih = (const float*)d_in[3];
  const float* bhh = (const float*)d_in[4];
  float* out = (float*)d_out;

  // Workspace layout (bytes), ~68 MB total
  char* ws = (char*)d_ws;
  _Float16* w16  = (_Float16*)(ws);                 // 33,554,432 B
  _Float16* x16  = (_Float16*)(ws + 33554432);      // 33,554,432 B
  _Float16* h16  = (_Float16*)(ws + 67108864);      // 262,144 B (2 parities)
  float*    bsum = (float*)   (ws + 67371008);      // 32,768 B
  float*    h32  = (float*)   (ws + 67403776);      // 262,144 B
  float*    c32  = (float*)   (ws + 67665920);      // 262,144 B

  cvt_weights_k<<<65536, 256, 0, stream>>>(wih, whh, w16);
  cvt_x_k      <<<65536, 256, 0, stream>>>(x, x16);
  prep_state_k <<<512,   256, 0, stream>>>(bih, bhh, bsum, h16, h32, c32);

  LstmArgs g;
  g.x16 = x16; g.w16 = w16; g.bsum = bsum;
  g.c32 = c32; g.h32 = h32; g.h16 = h16; g.out = out;

  // t = 0..512: launch t computes h0(t) and h1(t-1)
  for (int t = 0; t <= TT; ++t)
    lstm_pair_k<<<64, 128, 0, stream>>>(g, t);

  tail_k<<<512, 256, 0, stream>>>(out + (size_t)BATCH * TT * HID, h32, c32);
}
